// GCNContextSubgraphClassifier_26731876451141
// MI455X (gfx1250) — compile-verified, matrix operand order checked by reading, not run on hardware
//
#include <hip/hip_runtime.h>

typedef __attribute__((ext_vector_type(16))) _Float16 v16h;
typedef __attribute__((ext_vector_type(8)))  float    v8f;

#define NN 100000
#define NE 800000
#define NG 512
#define DH 128
#define NL 3

// ---------------- generic zero fill ----------------
__global__ void k_zero(float* __restrict__ p, int n) {
    int i = blockIdx.x * blockDim.x + threadIdx.x;
    if (i < n) p[i] = 0.f;
}

// ---------------- degree / norms ----------------
__global__ void k_deg(const int* __restrict__ ei, float* __restrict__ deg, int ne) {
    int e = blockIdx.x * blockDim.x + threadIdx.x;
    if (e < ne) atomicAdd(&deg[ei[ne + e]], 1.0f);   // dst row of edge_index
}

__global__ void k_deg_fin(const float* __restrict__ deg, float* __restrict__ dinv,
                          float* __restrict__ selfw, int n) {
    int i = blockIdx.x * blockDim.x + threadIdx.x;
    if (i < n) {
        float d = deg[i] + 1.0f;          // + self loop
        dinv[i]  = rsqrtf(d);
        selfw[i] = 1.0f / d;
    }
}

// ---------------- WMMA GEMM: C[M x 128] = A[M x 128] * W[128 x 128] ----------------
// block = 256 threads = 8 waves; each wave owns a 16-row strip; block covers 128 rows.

__device__ __forceinline__ v16h loadBfrag(const _Float16 (*__restrict__ Wt)[136],
                                          int n, int kb, int lhalf) {
    v16h b;
    #pragma unroll
    for (int j = 0; j < 8; ++j) {
        b[j]     = Wt[n][kb + lhalf * 8 + j];
        b[8 + j] = Wt[n][kb + 16 + lhalf * 8 + j];
    }
    return b;
}

__global__ void k_wmma_gemm128(const float* __restrict__ A, const float* __restrict__ W,
                               float* __restrict__ C, int M) {
    __shared__ _Float16 Wt[128][136];   // transposed W (Wt[n][k]), +8 pad kills bank conflicts

    int tid = threadIdx.x;
    for (int i = tid; i < 128 * 128; i += 256) {
        int k = i >> 7, n = i & 127;
        Wt[n][k] = (_Float16)W[i];      // W is row-major [k][n]
    }

    int wave  = tid >> 5;
    int lane  = tid & 31;
    int lhalf = lane >> 4;              // 0 or 1
    int lmod  = lane & 15;
    int row0  = blockIdx.x * 128 + wave * 16;
    int arow  = row0 + lmod;
    int crow  = arow < M ? arow : (M - 1);   // clamp: unconditional vector loads, store guarded
    const float4* ap = (const float4*)(A + (size_t)crow * DH);

    // Load the lane's full 128-float A row slice up-front as vector loads (overlaps LDS fill)
    v16h afrag[4];
    #pragma unroll
    for (int kt = 0; kt < 4; ++kt) {
        float4 q0 = ap[kt * 8 + lhalf * 2];          // K = kb + lhalf*8 + 0..3
        float4 q1 = ap[kt * 8 + lhalf * 2 + 1];      // K = kb + lhalf*8 + 4..7
        float4 q2 = ap[kt * 8 + 4 + lhalf * 2];      // K = kb + 16 + lhalf*8 + 0..3
        float4 q3 = ap[kt * 8 + 4 + lhalf * 2 + 1];  // K = kb + 16 + lhalf*8 + 4..7
        v16h a;
        a[0]  = (_Float16)q0.x; a[1]  = (_Float16)q0.y;
        a[2]  = (_Float16)q0.z; a[3]  = (_Float16)q0.w;
        a[4]  = (_Float16)q1.x; a[5]  = (_Float16)q1.y;
        a[6]  = (_Float16)q1.z; a[7]  = (_Float16)q1.w;
        a[8]  = (_Float16)q2.x; a[9]  = (_Float16)q2.y;
        a[10] = (_Float16)q2.z; a[11] = (_Float16)q2.w;
        a[12] = (_Float16)q3.x; a[13] = (_Float16)q3.y;
        a[14] = (_Float16)q3.z; a[15] = (_Float16)q3.w;
        afrag[kt] = a;
    }

    __syncthreads();

    v8f acc[8];
    #pragma unroll
    for (int t = 0; t < 8; ++t) acc[t] = {};

    // Software pipeline: keep exactly one B fragment in flight ahead of the WMMA
    // consuming the current one (low register pressure, overlaps LDS latency).
    v16h bcur = loadBfrag(Wt, lmod, 0, lhalf);
    #pragma unroll
    for (int kt = 0; kt < 4; ++kt) {
        int kb = kt * 32;
        #pragma unroll
        for (int nt = 0; nt < 8; ++nt) {
            v16h bnext;
            if (nt < 7) {
                bnext = loadBfrag(Wt, (nt + 1) * 16 + lmod, kb, lhalf);
            } else if (kt < 3) {
                bnext = loadBfrag(Wt, lmod, kb + 32, lhalf);
            }
            acc[nt] = __builtin_amdgcn_wmma_f32_16x16x32_f16(
                false, afrag[kt], false, bcur, (short)0, acc[nt], false, false);
            bcur = bnext;
        }
    }

    // C/D layout: VGPR v, lanes 0-15 -> M=v, lanes 16-31 -> M=8+v; N = lane%16
    if (row0 + 16 <= M) {
        // full tile fast path: unguarded coalesced stores
        #pragma unroll
        for (int nt = 0; nt < 8; ++nt) {
            #pragma unroll
            for (int v = 0; v < 8; ++v) {
                int r = row0 + v + 8 * lhalf;
                C[(size_t)r * DH + nt * 16 + lmod] = acc[nt][v];
            }
        }
    } else {
        #pragma unroll
        for (int nt = 0; nt < 8; ++nt) {
            #pragma unroll
            for (int v = 0; v < 8; ++v) {
                int r = row0 + v + 8 * lhalf;
                if (r < M) C[(size_t)r * DH + nt * 16 + lmod] = acc[nt][v];
            }
        }
    }
}

// ---------------- aggregation ----------------
// agg[i] = lin[i] * selfw[node]   (one thread per float4)
__global__ void k_init_agg(const float* __restrict__ lin, const float* __restrict__ selfw,
                           float* __restrict__ agg, int n) {
    int i = blockIdx.x * blockDim.x + threadIdx.x;
    if (i < n * 32) {
        int node = i >> 5;
        float4 v = ((const float4*)lin)[i];
        float  w = selfw[node];
        float4 o; o.x = v.x * w; o.y = v.y * w; o.z = v.z * w; o.w = v.w * w;
        ((float4*)agg)[i] = o;
    }
}

// one wave per edge, each lane scatters 4 features (L2-resident atomics)
__global__ void k_scatter(const int* __restrict__ ei, const float* __restrict__ dinv,
                          const float* __restrict__ lin, float* __restrict__ agg, int ne) {
    int t = blockIdx.x * blockDim.x + threadIdx.x;
    int e = t >> 5, lane = t & 31;
    if (e < ne) {
        int s = ei[e], d = ei[ne + e];
        float w = dinv[s] * dinv[d];
        float4 v = ((const float4*)(lin + (size_t)s * DH))[lane];
        float* dp = agg + (size_t)d * DH + lane * 4;
        atomicAdd(dp + 0, w * v.x);
        atomicAdd(dp + 1, w * v.y);
        atomicAdd(dp + 2, w * v.z);
        atomicAdd(dp + 3, w * v.w);
    }
}

// ---------------- batchnorm ----------------
__global__ void k_bn_stats(const float* __restrict__ h, float* __restrict__ sum,
                           float* __restrict__ sumsq, int n) {
    int f = threadIdx.x;                 // 128 threads/block
    float s = 0.f, s2 = 0.f;
    for (int r = blockIdx.x; r < n; r += gridDim.x) {
        float v = h[(size_t)r * DH + f];
        s += v; s2 += v * v;
    }
    atomicAdd(&sum[f], s);
    atomicAdd(&sumsq[f], s2);
}

__global__ void k_bn_fin(const float* __restrict__ sum, const float* __restrict__ sumsq,
                         const float* __restrict__ gamma, float* __restrict__ mu,
                         float* __restrict__ rstdg, int n) {
    int f = threadIdx.x;
    float m   = sum[f] / (float)n;
    float var = sumsq[f] / (float)n - m * m;
    mu[f]    = m;
    rstdg[f] = rsqrtf(var + 1e-5f) * gamma[f];
}

__global__ void k_bn_apply(const float* __restrict__ agg, const float* __restrict__ mu,
                           const float* __restrict__ rstdg, const float* __restrict__ beta,
                           float* __restrict__ out, int n) {
    int i = blockIdx.x * blockDim.x + threadIdx.x;
    if (i < n * 32) {
        int f0 = (i & 31) * 4;
        float4 v = ((const float4*)agg)[i];
        float4 o;
        o.x = fmaxf(0.f, (v.x - mu[f0 + 0]) * rstdg[f0 + 0] + beta[f0 + 0]);
        o.y = fmaxf(0.f, (v.y - mu[f0 + 1]) * rstdg[f0 + 1] + beta[f0 + 1]);
        o.z = fmaxf(0.f, (v.z - mu[f0 + 2]) * rstdg[f0 + 2] + beta[f0 + 2]);
        o.w = fmaxf(0.f, (v.w - mu[f0 + 3]) * rstdg[f0 + 3] + beta[f0 + 3]);
        ((float4*)out)[i] = o;
    }
}

// ---------------- pooling ----------------
__global__ void k_pool(const float* __restrict__ h, const int* __restrict__ batch,
                       float* __restrict__ pool, int n) {
    int i = blockIdx.x * blockDim.x + threadIdx.x;
    if (i < n * DH) {
        int node = i >> 7, f = i & 127;
        atomicAdd(&pool[(size_t)batch[node] * DH + f], h[i]);
    }
}

__global__ void k_cnt(const int* __restrict__ batch, float* __restrict__ cnt, int n) {
    int i = blockIdx.x * blockDim.x + threadIdx.x;
    if (i < n) atomicAdd(&cnt[batch[i]], 1.0f);
}

__global__ void k_pool_div(float* __restrict__ pool, const float* __restrict__ cnt, int g) {
    int i = blockIdx.x * blockDim.x + threadIdx.x;
    if (i < g * DH) pool[i] /= fmaxf(cnt[i >> 7], 1.0f);
}

// ---------------- MLP head ----------------
__global__ void k_bias_relu(float* __restrict__ hz, const float* __restrict__ b, int n) {
    int i = blockIdx.x * blockDim.x + threadIdx.x;
    if (i < n) hz[i] = fmaxf(0.f, hz[i] + b[i & 127]);
}

// one wave per graph: 128-dot with W2, wave reduction
__global__ void k_logits(const float* __restrict__ hz, const float* __restrict__ W2,
                         const float* __restrict__ b2, float* __restrict__ out, int g) {
    int t = blockIdx.x * blockDim.x + threadIdx.x;
    int gi = t >> 5, lane = t & 31;
    if (gi < g) {
        const float* hp = hz + (size_t)gi * DH;
        float s = 0.f;
        #pragma unroll
        for (int j = 0; j < 4; ++j) s += hp[lane * 4 + j] * W2[lane * 4 + j];
        #pragma unroll
        for (int m = 16; m >= 1; m >>= 1) s += __shfl_xor(s, m, 32);
        if (lane == 0) out[gi] = s + b2[0];
    }
}

extern "C" void kernel_launch(void* const* d_in, const int* in_sizes, int n_in,
                              void* d_out, int out_size, void* d_ws, size_t ws_size,
                              hipStream_t stream) {
    const float* x      = (const float*)d_in[0];
    const int*   eidx   = (const int*)  d_in[1];
    const int*   batch  = (const int*)  d_in[2];
    const float* Ws     = (const float*)d_in[3];
    // d_in[4] = bs: skipped — per-feature bias is absorbed exactly by BatchNorm
    const float* gammas = (const float*)d_in[5];
    const float* betas  = (const float*)d_in[6];
    const float* W1     = (const float*)d_in[7];
    const float* b1     = (const float*)d_in[8];
    const float* W2     = (const float*)d_in[9];
    const float* b2     = (const float*)d_in[10];
    float* out = (float*)d_out;

    // workspace layout (floats)
    float* ws    = (float*)d_ws;
    float* DEG   = ws;                         // NN
    float* DINV  = DEG   + NN;                 // NN
    float* SELFW = DINV  + NN;                 // NN
    float* H     = SELFW + NN;                 // NN*DH
    float* LIN   = H     + (size_t)NN * DH;    // NN*DH
    float* AGG   = LIN   + (size_t)NN * DH;    // NN*DH
    float* SUM   = AGG   + (size_t)NN * DH;    // DH
    float* SUMSQ = SUM   + DH;                 // DH
    float* MU    = SUMSQ + DH;                 // DH
    float* RSTDG = MU    + DH;                 // DH
    float* POOL  = RSTDG + DH;                 // NG*DH
    float* CNT   = POOL  + (size_t)NG * DH;    // NG
    float* HZ    = CNT   + NG;                 // NG*DH

    const int B = 256;

    // degree + norms
    k_zero<<<(NN + B - 1) / B, B, 0, stream>>>(DEG, NN);
    k_deg<<<(NE + B - 1) / B, B, 0, stream>>>(eidx, DEG, NE);
    k_deg_fin<<<(NN + B - 1) / B, B, 0, stream>>>(DEG, DINV, SELFW, NN);

    // 3 GCN layers
    for (int l = 0; l < NL; ++l) {
        const float* hin = (l == 0) ? x : H;
        const float* Wl  = Ws + (size_t)l * DH * DH;
        k_wmma_gemm128<<<(NN + 127) / 128, 256, 0, stream>>>(hin, Wl, LIN, NN);
        k_init_agg<<<(NN * 32 + B - 1) / B, B, 0, stream>>>(LIN, SELFW, AGG, NN);
        k_scatter<<<(NE * 32 + B - 1) / B, B, 0, stream>>>(eidx, DINV, LIN, AGG, NE);
        k_zero<<<1, 256, 0, stream>>>(SUM, 2 * DH);      // SUM + SUMSQ contiguous
        k_bn_stats<<<1024, 128, 0, stream>>>(AGG, SUM, SUMSQ, NN);
        k_bn_fin<<<1, 128, 0, stream>>>(SUM, SUMSQ, gammas + (size_t)l * DH, MU, RSTDG, NN);
        k_bn_apply<<<(NN * 32 + B - 1) / B, B, 0, stream>>>(AGG, MU, RSTDG,
                                                            betas + (size_t)l * DH, H, NN);
    }

    // global mean pool
    k_zero<<<(NG * DH + NG + B - 1) / B, B, 0, stream>>>(POOL, NG * DH + NG); // POOL+CNT contiguous
    k_pool<<<((size_t)NN * DH + B - 1) / B, B, 0, stream>>>(H, batch, POOL, NN);
    k_cnt<<<(NN + B - 1) / B, B, 0, stream>>>(batch, CNT, NN);
    k_pool_div<<<(NG * DH + B - 1) / B, B, 0, stream>>>(POOL, CNT, NG);

    // MLP head
    k_wmma_gemm128<<<(NG + 127) / 128, 256, 0, stream>>>(POOL, W1, HZ, NG);
    k_bias_relu<<<(NG * DH + B - 1) / B, B, 0, stream>>>(HZ, b1, NG * DH);
    k_logits<<<(NG * 32 + B - 1) / B, B, 0, stream>>>(HZ, W2, b2, out, NG);
}
